// AttentionBlock_81123342287029
// MI455X (gfx1250) — compile-verified, matrix operand order checked by reading, not run on hardware
//
#include <hip/hip_runtime.h>
#include <hip/hip_bf16.h>

typedef __attribute__((ext_vector_type(16))) _Float16 v16h;
typedef __attribute__((ext_vector_type(8)))  float    v8f;
typedef __attribute__((ext_vector_type(4)))  _Float16 h4;

// ---------------------------------------------------------------------------
// Problem constants
// ---------------------------------------------------------------------------
static constexpr int BATCH   = 2;
static constexpr int SEQ     = 4096;
static constexpr int HIDDEN  = 1024;
static constexpr int HEADS   = 16;
static constexpr int HEAD_D  = 64;
static constexpr int THREEC  = 3 * HIDDEN;   // 3072
static constexpr int MTOT    = BATCH * SEQ;  // 8192

// ---------------------------------------------------------------------------
// CDNA5 async global->LDS (16B per lane, ASYNCcnt-tracked)
// ---------------------------------------------------------------------------
static __device__ __forceinline__ void async_copy_b128(const _Float16* gsrc,
                                                       const _Float16* ldst) {
  const unsigned int       ldsa = (unsigned int)(uintptr_t)ldst;
  const unsigned long long ga   = (unsigned long long)(uintptr_t)gsrc;
  asm volatile("global_load_async_to_lds_b128 %0, %1, off"
               :: "v"(ldsa), "v"(ga) : "memory");
}
static __device__ __forceinline__ void wait_async0() {
  asm volatile("s_wait_asynccnt 0" ::: "memory");
}

// ---------------------------------------------------------------------------
// WMMA helper (D = A(16x32 f16) * B(32x16 f16) + C(16x16 f32))
// ---------------------------------------------------------------------------
static __device__ __forceinline__ v8f wmma32(v16h a, v16h b, v8f c) {
  return __builtin_amdgcn_wmma_f32_16x16x32_f16(false, a, false, b,
                                                (short)0, c, false, false);
}

// A-fragment: 16x32 f16 from LDS. Lanes 0-15 -> M=lane, K 0-7 and 16-23;
// lanes 16-31 -> K 8-15 and 24-31. 8 contiguous halves per chunk -> ds_load_b128.
static __device__ __forceinline__ v16h lds_a_frag(const _Float16* base,
                                                  int row0, int ld, int k0) {
  const int lane = threadIdx.x & 31;
  const int m = lane & 15, hf = lane >> 4;
  v16h a;
#pragma unroll
  for (int v = 0; v < 8; ++v) {
    const int kb = (v < 4) ? (hf * 8 + 2 * v) : (16 + hf * 8 + 2 * (v - 4));
    const _Float16* p = base + (size_t)(row0 + m) * ld + k0 + kb;
    a[2 * v]     = p[0];
    a[2 * v + 1] = p[1];
  }
  return a;
}

// B-fragment (32x16, K x N) from an N-major LDS tile: elem B[k][n] = base[n*ld + k]
static __device__ __forceinline__ v16h lds_b_frag_nmajor(const _Float16* base,
                                                         int n0, int ld, int k0) {
  const int lane = threadIdx.x & 31;
  const int n = lane & 15, hf = lane >> 4;
  v16h b;
#pragma unroll
  for (int v = 0; v < 8; ++v) {
    const int kb = (v < 4) ? (hf * 8 + 2 * v) : (16 + hf * 8 + 2 * (v - 4));
    const _Float16* p = base + (size_t)(n0 + n) * ld + k0 + kb;
    b[2 * v]     = p[0];
    b[2 * v + 1] = p[1];
  }
  return b;
}

// ---------------------------------------------------------------------------
// f32 -> f16 bulk conversion (vectorized, one-shot preprocessing)
// ---------------------------------------------------------------------------
__global__ __launch_bounds__(256)
void cvt_f32_f16(const float* __restrict__ src, _Float16* __restrict__ dst,
                 int n4) {
  const int i = blockIdx.x * 256 + threadIdx.x;
  if (i < n4) {
    const float4 v = ((const float4*)src)[i];
    h4 o; o[0] = (_Float16)v.x; o[1] = (_Float16)v.y;
          o[2] = (_Float16)v.z; o[3] = (_Float16)v.w;
    ((h4*)dst)[i] = o;
  }
}

// ---------------------------------------------------------------------------
// GEMM: out[M, Nout] = A[M, K] * W[Nout, K]^T + bias, A/W f16.
//   128(M) x 128(N) x 64(K) tile; 256 threads = 8 waves (4x2); each wave
//   computes 32x64 = 2x4 WMMA tiles -> 16 WMMA per K-iter between barriers.
//   Double-buffered LDS (64 KB), all staging via async global->LDS DMA,
//   pipelined one K-block ahead.
// ---------------------------------------------------------------------------
template <typename OT>
__global__ __launch_bounds__(256)
void gemm_bias_wmma(const _Float16* __restrict__ A, const _Float16* __restrict__ W,
                    const float* __restrict__ bias, OT* __restrict__ out,
                    int M, int Nout, int K) {
  constexpr int BM = 128, BN = 128, BK = 64;
  __shared__ _Float16 As[2][BM * BK];  // 2 x 16 KB
  __shared__ _Float16 Ws[2][BN * BK];  // 2 x 16 KB

  const int tid   = threadIdx.x;
  const int m_blk = blockIdx.y * BM;
  const int n_blk = blockIdx.x * BN;
  const int w     = tid >> 5;
  const int lane  = tid & 31;
  const int wm    = w >> 1;   // 0..3 -> 32 rows each
  const int wn    = w & 1;    // 0..1 -> 64 cols each

  // Per-thread staging chunk coordinates (8 halves = 16 B per chunk)
  int rs[4], cs[4];
  const _Float16* pa[4];  // global A pointers, advanced by BK each K-block
  const _Float16* pw[4];  // global W pointers
#pragma unroll
  for (int i = 0; i < 4; ++i) {
    const int idx = tid + i * 256;
    rs[i] = idx >> 3; cs[i] = (idx & 7) * 8;   // rows 0..127, cols {0,8,..,56}
    pa[i] = A + (size_t)(m_blk + rs[i]) * K + cs[i];
    pw[i] = W + (size_t)(n_blk + rs[i]) * K + cs[i];
  }

  // Prologue: stage K-block 0 into buffer 0
#pragma unroll
  for (int i = 0; i < 4; ++i) {
    async_copy_b128(pa[i], &As[0][rs[i] * BK + cs[i]]);
    async_copy_b128(pw[i], &Ws[0][rs[i] * BK + cs[i]]);
    pa[i] += BK;
    pw[i] += BK;
  }

  v8f acc[2][4] = {};

  const int nkb = K / BK;
  for (int kb = 0; kb < nkb; ++kb) {
    const int buf = kb & 1;
    wait_async0();
    __syncthreads();

    if (kb + 1 < nkb) {  // stage next K-block into other buffer
      const int nb = buf ^ 1;
#pragma unroll
      for (int i = 0; i < 4; ++i) {
        async_copy_b128(pa[i], &As[nb][rs[i] * BK + cs[i]]);
        async_copy_b128(pw[i], &Ws[nb][rs[i] * BK + cs[i]]);
        pa[i] += BK;
        pw[i] += BK;
      }
    }

#pragma unroll
    for (int ks = 0; ks < BK; ks += 32) {
      v16h af[2], bf[4];
#pragma unroll
      for (int mt = 0; mt < 2; ++mt)
        af[mt] = lds_a_frag(As[buf], wm * 32 + mt * 16, BK, ks);
#pragma unroll
      for (int nt = 0; nt < 4; ++nt)
        bf[nt] = lds_b_frag_nmajor(Ws[buf], wn * 64 + nt * 16, BK, ks);
#pragma unroll
      for (int mt = 0; mt < 2; ++mt)
#pragma unroll
        for (int nt = 0; nt < 4; ++nt)
          acc[mt][nt] = wmma32(af[mt], bf[nt], acc[mt][nt]);
    }
  }

  // Epilogue: C/D layout -> lane n = lane&15; row = r + 8*(lane>=16)
  const int hf = lane >> 4, nin = lane & 15;
#pragma unroll
  for (int mt = 0; mt < 2; ++mt)
#pragma unroll
    for (int nt = 0; nt < 4; ++nt) {
      const int nglob = n_blk + wn * 64 + nt * 16 + nin;
      const float bv = bias[nglob];
#pragma unroll
      for (int r = 0; r < 8; ++r) {
        const int mglob = m_blk + wm * 32 + mt * 16 + r + hf * 8;
        out[(size_t)mglob * Nout + nglob] = (OT)(acc[mt][nt][r] + bv);
      }
    }
}

// ---------------------------------------------------------------------------
// Flash attention: one block = one (b, h, 64-query slab). 128 threads = 4 waves,
// wave w owns query rows [w*16, w*16+16). Loop over 64-wide key blocks.
//   - Q and K staged with async global->LDS DMA; K double-buffered and
//     prefetched one block ahead (DMA overlaps S/softmax/PV compute).
//   - V prefetched one block ahead into registers, transposed (d-major) into
//     LDS so P*V B-fragments are contiguous (ds_load_b128).
// qkv layout: [B, N, 3C] f16 ; out layout: [B, N, H, D] f16
// ---------------------------------------------------------------------------
__global__ __launch_bounds__(128)
void flash_attn_wmma(const _Float16* __restrict__ qkv,
                     _Float16* __restrict__ attn_out) {
  constexpr int BR = 64, BC = 64, DH = HEAD_D;
  constexpr int SLD = 65;  // padded f32 score stride (bank-conflict free)
  constexpr int PLD = 72;  // padded f16 P stride (keeps 16B-aligned chunks)
  constexpr int VLD = 72;  // padded f16 V^T stride

  __shared__ _Float16 Qs[BR * DH];        //  8 KB
  __shared__ _Float16 Ks[2][BC * DH];     // 16 KB (double-buffered)
  __shared__ _Float16 VsT[DH * VLD];      //  9 KB (d-major)
  __shared__ float    Ss[BR * SLD];       // 16.25 KB
  __shared__ _Float16 Ps[BR * PLD];       //  9 KB
  __shared__ float    m_s[BR], l_s[BR], alpha_s[BR];

  const int tid  = threadIdx.x;
  const int w    = tid >> 5;
  const int lane = tid & 31;
  const int hf   = lane >> 4, nin = lane & 15;
  const int qblk = blockIdx.x;
  const int h    = blockIdx.y;
  const int b    = blockIdx.z;
  const float scale = 0.125f;             // 1/sqrt(64)

  const size_t rowstride = (size_t)THREEC;
  const size_t blkstep   = (size_t)BC * rowstride;   // halves per key-block
  const _Float16* qbase = qkv + ((size_t)b * SEQ) * rowstride + h * HEAD_D;

  // --- Prologue: async-stage Q slab and K block 0; V block 0 into registers ---
  int rk[4], ck[4];
  const _Float16* gk[4];
#pragma unroll
  for (int i = 0; i < 4; ++i) {
    const int idx = tid + i * 128;
    rk[i] = idx >> 3; ck[i] = (idx & 7) * 8;
    // Q slab (same chunk shape)
    async_copy_b128(qbase + (size_t)(qblk * BR + rk[i]) * rowstride + ck[i],
                    &Qs[rk[i] * DH + ck[i]]);
    gk[i] = qbase + (size_t)rk[i] * rowstride + HIDDEN + ck[i];
    async_copy_b128(gk[i], &Ks[0][rk[i] * DH + ck[i]]);
    gk[i] += blkstep;
  }

  int rv[8], cv[8];
  const _Float16* gv[8];
  h4 vreg[8];
#pragma unroll
  for (int i = 0; i < 8; ++i) {
    const int idx = tid + i * 128;
    rv[i] = idx >> 4; cv[i] = (idx & 15) * 4;
    gv[i] = qbase + (size_t)rv[i] * rowstride + 2 * HIDDEN + cv[i];
    vreg[i] = *(const h4*)gv[i];
    gv[i] += blkstep;
  }

  if (tid < BR) { m_s[tid] = -3.0e38f; l_s[tid] = 0.0f; }

  v8f oacc[4] = {};  // 16 rows x 64 cols f32, per wave

  const int nblocks = SEQ / BC;
  for (int kb = 0; kb < nblocks; ++kb) {
    const int buf = kb & 1;
    wait_async0();
    __syncthreads();   // Ks[buf]/Qs ready; prev PV done reading VsT/Ps

    // Commit V block kb (registers -> transposed LDS)
#pragma unroll
    for (int i = 0; i < 8; ++i)
#pragma unroll
      for (int j = 0; j < 4; ++j) VsT[(cv[i] + j) * VLD + rv[i]] = vreg[i][j];

    if (kb + 1 < nblocks) {
      // Async-prefetch K block kb+1 into the other buffer
#pragma unroll
      for (int i = 0; i < 4; ++i) {
        async_copy_b128(gk[i], &Ks[buf ^ 1][rk[i] * DH + ck[i]]);
        gk[i] += blkstep;
      }
      // Prefetch V block kb+1 into registers
#pragma unroll
      for (int i = 0; i < 8; ++i) {
        vreg[i] = *(const h4*)gv[i];
        gv[i] += blkstep;
      }
    }

    // S = Q K^T (16x64 per wave): 4 n-tiles, K-dim = 64 -> 2 WMMA steps
    v8f sacc[4] = {};
#pragma unroll
    for (int ks = 0; ks < 2; ++ks) {
      const v16h af = lds_a_frag(Qs, w * 16, DH, ks * 32);
#pragma unroll
      for (int nt = 0; nt < 4; ++nt) {
        const v16h bf = lds_b_frag_nmajor(Ks[buf], nt * 16, DH, ks * 32);
        sacc[nt] = wmma32(af, bf, sacc[nt]);
      }
    }
#pragma unroll
    for (int nt = 0; nt < 4; ++nt)
#pragma unroll
      for (int r = 0; r < 8; ++r)
        Ss[(w * 16 + r + hf * 8) * SLD + nt * 16 + nin] = sacc[nt][r] * scale;
    __syncthreads();

    // Online softmax: thread t owns row t (t < 64)
    if (tid < BR) {
      const float* srow = &Ss[tid * SLD];
      float rmax = -3.0e38f;
#pragma unroll 8
      for (int j = 0; j < BC; ++j) rmax = fmaxf(rmax, srow[j]);
      const float m_old = m_s[tid];
      const float m_new = fmaxf(m_old, rmax);
      const float alpha = __expf(m_old - m_new);
      float psum = 0.0f;
      _Float16* prow = &Ps[tid * PLD];
#pragma unroll 8
      for (int j = 0; j < BC; ++j) {
        const float p = __expf(srow[j] - m_new);
        prow[j] = (_Float16)p;
        psum += p;
      }
      l_s[tid]     = l_s[tid] * alpha + psum;
      m_s[tid]     = m_new;
      alpha_s[tid] = alpha;
    }
    __syncthreads();

    // Rescale accumulators, then O += P * V (4 d-tiles, 2 WMMA K-steps)
#pragma unroll
    for (int r = 0; r < 8; ++r) {
      const float a = alpha_s[w * 16 + r + hf * 8];
#pragma unroll
      for (int dt = 0; dt < 4; ++dt) oacc[dt][r] *= a;
    }
#pragma unroll
    for (int ks = 0; ks < 2; ++ks) {
      const v16h af = lds_a_frag(Ps, w * 16, PLD, ks * 32);
#pragma unroll
      for (int dt = 0; dt < 4; ++dt) {
        const v16h bf = lds_b_frag_nmajor(VsT, dt * 16, VLD, ks * 32);
        oacc[dt] = wmma32(af, bf, oacc[dt]);
      }
    }
  }
  __syncthreads();

  // Write O / l  ->  attn_out[b, nq, h, d]  (f16)
#pragma unroll
  for (int r = 0; r < 8; ++r) {
    const int mloc = w * 16 + r + hf * 8;
    const float linv = 1.0f / l_s[mloc];
    const int nq = qblk * BR + mloc;
    _Float16* dst = attn_out + ((size_t)b * SEQ + nq) * HIDDEN + h * HEAD_D;
#pragma unroll
    for (int dt = 0; dt < 4; ++dt)
      dst[dt * 16 + nin] = (_Float16)(oacc[dt][r] * linv);
  }
}

// ---------------------------------------------------------------------------
// Host-side launch
// ---------------------------------------------------------------------------
extern "C" void kernel_launch(void* const* d_in, const int* in_sizes, int n_in,
                              void* d_out, int out_size, void* d_ws, size_t ws_size,
                              hipStream_t stream) {
  (void)in_sizes; (void)n_in; (void)out_size; (void)ws_size;
  const float* x     = (const float*)d_in[0];   // [2, 4096, 1024]
  const float* qkv_w = (const float*)d_in[1];   // [3072, 1024]
  const float* qkv_b = (const float*)d_in[2];   // [3072]
  const float* out_w = (const float*)d_in[3];   // [1024, 1024]
  const float* out_b = (const float*)d_in[4];   // [1024]
  float* out = (float*)d_out;                   // [2, 4096, 1024]

  // Workspace layout (all f16 halves)
  _Float16* qkv_h  = (_Float16*)d_ws;                     // 8192 x 3072
  _Float16* attn_h = qkv_h  + (size_t)MTOT * THREEC;      // 8192 x 1024
  _Float16* x_h    = attn_h + (size_t)MTOT * HIDDEN;      // 8192 x 1024
  _Float16* qkvw_h = x_h    + (size_t)MTOT * HIDDEN;      // 3072 x 1024
  _Float16* outw_h = qkvw_h + (size_t)THREEC * HIDDEN;    // 1024 x 1024

  // 0) One-shot f32 -> f16 conversion of activations and weights
  {
    const int n4x = (MTOT * HIDDEN) / 4;
    cvt_f32_f16<<<n4x / 256, 256, 0, stream>>>(x, x_h, n4x);
    const int n4q = (THREEC * HIDDEN) / 4;
    cvt_f32_f16<<<n4q / 256, 256, 0, stream>>>(qkv_w, qkvw_h, n4q);
    const int n4o = (HIDDEN * HIDDEN) / 4;
    cvt_f32_f16<<<n4o / 256, 256, 0, stream>>>(out_w, outw_h, n4o);
  }

  // 1) QKV projection: [8192,1024] x [1024,3072]^T + bias -> f16
  gemm_bias_wmma<_Float16>
      <<<dim3(THREEC / 128, MTOT / 128), 256, 0, stream>>>(
          x_h, qkvw_h, qkv_b, qkv_h, MTOT, THREEC, HIDDEN);

  // 2) Flash attention over heads
  flash_attn_wmma<<<dim3(SEQ / 64, HEADS, BATCH), 128, 0, stream>>>(qkv_h, attn_h);

  // 3) Output projection: [8192,1024] x [1024,1024]^T + bias -> f32
  gemm_bias_wmma<float>
      <<<dim3(HIDDEN / 128, MTOT / 128), 256, 0, stream>>>(
          attn_h, outw_h, out_b, out, MTOT, HIDDEN, HIDDEN);
}
